// QLinear_25031069401194
// MI455X (gfx1250) — 8-bit matrix operand order repaired and checked by reading, compiled, not run on hardware
//
#include <hip/hip_runtime.h>
#include <hip/hip_bf16.h>

typedef __attribute__((ext_vector_type(8))) int v8i;

#define M_DIM 2048
#define K_DIM 4096
#define N_DIM 11008

#define BM 128
#define BN 128
#define BK 64

#define FBM 64
#define FBN 128

__device__ __forceinline__ void async_copy_b128(void* lds_dst, const void* gsrc) {
    unsigned lds = (unsigned)(size_t)lds_dst;
    unsigned long long g = (unsigned long long)gsrc;
    asm volatile("global_load_async_to_lds_b128 %0, %1, off"
                 :: "v"(lds), "v"(g) : "memory");
}
__device__ __forceinline__ void wait_async_zero() {
    asm volatile("s_wait_asynccnt 0x0" ::: "memory");
}

__global__ __launch_bounds__(256)
void absmax_kernel(const float* __restrict__ x, unsigned int* __restrict__ amax, int n) {
    int idx = blockIdx.x * blockDim.x + threadIdx.x;
    int stride = gridDim.x * blockDim.x;
    int n4 = n >> 2;
    const float4* x4 = (const float4*)x;
    float m = 0.0f;
    for (int i = idx; i < n4; i += stride) {
        float4 v = x4[i];
        m = fmaxf(m, fmaxf(fmaxf(fabsf(v.x), fabsf(v.y)),
                           fmaxf(fabsf(v.z), fabsf(v.w))));
    }
    #pragma unroll
    for (int off = 16; off > 0; off >>= 1)
        m = fmaxf(m, __shfl_xor(m, off, 32));
    __shared__ float smax[8];
    int lane = threadIdx.x & 31;
    int wv   = threadIdx.x >> 5;
    if (lane == 0) smax[wv] = m;
    __syncthreads();
    if (wv == 0) {
        m = (lane < 8) ? smax[lane] : 0.0f;
        #pragma unroll
        for (int off = 4; off > 0; off >>= 1)
            m = fmaxf(m, __shfl_xor(m, off, 32));
        if (lane == 0) atomicMax(amax, __float_as_uint(m));
    }
}

__global__ __launch_bounds__(256)
void quant_kernel(const float* __restrict__ x, int* __restrict__ xq_words,
                  const unsigned int* __restrict__ amax, int n) {
    float scale = __uint_as_float(*amax) * (1.0f / 128.0f);
    float inv = (scale > 0.0f) ? (1.0f / scale) : 0.0f;
    int idx = blockIdx.x * blockDim.x + threadIdx.x;
    int stride = gridDim.x * blockDim.x;
    int n4 = n >> 2;
    const float4* x4 = (const float4*)x;
    for (int i = idx; i < n4; i += stride) {
        float4 v = x4[i];
        int a = (int)rintf(v.x * inv); a = min(max(a, -128), 127);
        int b = (int)rintf(v.y * inv); b = min(max(b, -128), 127);
        int c = (int)rintf(v.z * inv); c = min(max(c, -128), 127);
        int d = (int)rintf(v.w * inv); d = min(max(d, -128), 127);
        xq_words[i] = (a & 0xff) | ((b & 0xff) << 8) | ((c & 0xff) << 16) | ((d & 0xff) << 24);
    }
}

__global__ __launch_bounds__(256)
void wtrans_kernel(const signed char* __restrict__ W, signed char* __restrict__ Wt) {
    __shared__ __align__(16) signed char T[128][64];
    const int tid = threadIdx.x;
    const int n0 = blockIdx.x * 128;
    const int k0 = blockIdx.y * 64;
    #pragma unroll
    for (int i = 0; i < 8; ++i) {
        int w  = tid + i * 256;
        int kk = w >> 5;
        int nn = (w & 31) << 2;
        int word = *(const int*)(W + (size_t)(k0 + kk) * N_DIM + (n0 + nn));
        T[nn + 0][kk] = (signed char)( word        & 0xff);
        T[nn + 1][kk] = (signed char)((word >> 8)  & 0xff);
        T[nn + 2][kk] = (signed char)((word >> 16) & 0xff);
        T[nn + 3][kk] = (signed char)((word >> 24) & 0xff);
    }
    __syncthreads();
    #pragma unroll
    for (int i = 0; i < 2; ++i) {
        int t = tid + i * 256;
        int r = t >> 2;
        int c = (t & 3) << 4;
        *(int4*)(Wt + (size_t)(n0 + r) * K_DIM + k0 + c) = *(const int4*)&T[r][c];
    }
}

__global__ __launch_bounds__(256)
void gemm_i8_t_kernel(const signed char* __restrict__ Aq,
                      const signed char* __restrict__ Bt,
                      const float* __restrict__ bias,
                      const float* __restrict__ y_scale,
                      const unsigned int* __restrict__ amax_bits,
                      float* __restrict__ Y) {
    __shared__ __align__(16) signed char As[2][BM][BK];
    __shared__ __align__(16) signed char Bs[2][BN][BK];

    const int tid  = threadIdx.x;
    const int lane = tid & 31;
    const int wave = tid >> 5;
    const int wm   = wave >> 2;
    const int wn   = wave & 3;
    const int half = lane >> 4;
    const int l15  = lane & 15;

    const int block_m = blockIdx.y * BM;
    const int block_n = blockIdx.x * BN;

    const int sr = tid >> 2;
    const int sc = (tid & 3) << 4;

    const signed char* a_src0 = Aq + (size_t)(block_m + sr)      * K_DIM + sc;
    const signed char* a_src1 = Aq + (size_t)(block_m + sr + 64) * K_DIM + sc;
    const signed char* b_src0 = Bt + (size_t)(block_n + sr)      * K_DIM + sc;
    const signed char* b_src1 = Bt + (size_t)(block_n + sr + 64) * K_DIM + sc;

    v8i acc[4][2] = {};

    async_copy_b128(&As[0][sr][sc],      a_src0);
    async_copy_b128(&As[0][sr + 64][sc], a_src1);
    async_copy_b128(&Bs[0][sr][sc],      b_src0);
    async_copy_b128(&Bs[0][sr + 64][sc], b_src1);

    const int NSTEP = K_DIM / BK;
    for (int s = 0; s < NSTEP; ++s) {
        const int cur = s & 1;
        wait_async_zero();
        __syncthreads();

        if (s + 1 < NSTEP) {
            int k1 = (s + 1) * BK;
            async_copy_b128(&As[cur ^ 1][sr][sc],      a_src0 + k1);
            async_copy_b128(&As[cur ^ 1][sr + 64][sc], a_src1 + k1);
            async_copy_b128(&Bs[cur ^ 1][sr][sc],      b_src0 + k1);
            async_copy_b128(&Bs[cur ^ 1][sr + 64][sc], b_src1 + k1);
        }

        v8i afrag[4], bfrag[2];
        #pragma unroll
        for (int mt = 0; mt < 4; ++mt) {
            int m = wm * 64 + mt * 16 + l15;
            #pragma unroll
            for (int j = 0; j < 8; ++j) {
                int kk = ((j >> 2) << 5) + (half << 4) + ((j & 3) << 2);
                afrag[mt][j] = *(const int*)&As[cur][m][kk];
            }
        }
        #pragma unroll
        for (int nt = 0; nt < 2; ++nt) {
            int n = wn * 32 + nt * 16 + l15;
            #pragma unroll
            for (int j = 0; j < 8; ++j) {
                int kk = ((j >> 2) << 5) + (half << 4) + ((j & 3) << 2);
                bfrag[nt][j] = *(const int*)&Bs[cur][n][kk];
            }
        }

        #pragma unroll
        for (int mt = 0; mt < 4; ++mt)
            #pragma unroll
            for (int nt = 0; nt < 2; ++nt)
                acc[mt][nt] = __builtin_amdgcn_wmma_i32_16x16x64_iu8(
                    true, afrag[mt], true, bfrag[nt], acc[mt][nt], false, false);
    }

    float xs = __uint_as_float(*amax_bits) * (1.0f / 128.0f);
    float sf = xs * y_scale[0];
    #pragma unroll
    for (int mt = 0; mt < 4; ++mt) {
        #pragma unroll
        for (int nt = 0; nt < 2; ++nt) {
            int n = block_n + wn * 32 + nt * 16 + l15;
            float bv = bias[n];
            #pragma unroll
            for (int r = 0; r < 8; ++r) {
                int m = block_m + wm * 64 + mt * 16 + half * 8 + r;
                Y[(size_t)m * N_DIM + n] = (float)acc[mt][nt][r] * sf + bv;
            }
        }
    }
}

__global__ __launch_bounds__(256)
void gemm_i8_kernel(const signed char* __restrict__ Aq,
                    const signed char* __restrict__ Bq,
                    const float* __restrict__ bias,
                    const float* __restrict__ y_scale,
                    const unsigned int* __restrict__ amax_bits,
                    float* __restrict__ Y) {
    __shared__ __align__(16) signed char As[FBM][BK];
    __shared__ __align__(16) signed char BsT[FBN][BK];

    const int tid  = threadIdx.x;
    const int lane = tid & 31;
    const int wave = tid >> 5;
    const int wm   = wave >> 2;
    const int wn   = wave & 3;
    const int half = lane >> 4;
    const int l15  = lane & 15;
    const int block_m = blockIdx.y * FBM;
    const int block_n = blockIdx.x * FBN;

    v8i acc[2][2] = {};

    for (int k0 = 0; k0 < K_DIM; k0 += BK) {
        {
            int r = tid >> 2, c = (tid & 3) << 4;
            *(int4*)&As[r][c] = *(const int4*)(Aq + (size_t)(block_m + r) * K_DIM + k0 + c);
        }
        #pragma unroll
        for (int i = 0; i < 8; ++i) {
            int w = tid + i * 256, kk = w >> 5, nn = (w & 31) << 2;
            int word = *(const int*)(Bq + (size_t)(k0 + kk) * N_DIM + block_n + nn);
            BsT[nn + 0][kk] = (signed char)( word        & 0xff);
            BsT[nn + 1][kk] = (signed char)((word >> 8)  & 0xff);
            BsT[nn + 2][kk] = (signed char)((word >> 16) & 0xff);
            BsT[nn + 3][kk] = (signed char)((word >> 24) & 0xff);
        }
        __syncthreads();

        v8i afrag[2], bfrag[2];
        #pragma unroll
        for (int mt = 0; mt < 2; ++mt) {
            int m = wm * 32 + mt * 16 + l15;
            #pragma unroll
            for (int j = 0; j < 8; ++j)
                afrag[mt][j] = *(const int*)&As[m][((j >> 2) << 5) + (half << 4) + ((j & 3) << 2)];
        }
        #pragma unroll
        for (int nt = 0; nt < 2; ++nt) {
            int n = wn * 32 + nt * 16 + l15;
            #pragma unroll
            for (int j = 0; j < 8; ++j)
                bfrag[nt][j] = *(const int*)&BsT[n][((j >> 2) << 5) + (half << 4) + ((j & 3) << 2)];
        }
        #pragma unroll
        for (int mt = 0; mt < 2; ++mt)
            #pragma unroll
            for (int nt = 0; nt < 2; ++nt)
                acc[mt][nt] = __builtin_amdgcn_wmma_i32_16x16x64_iu8(
                    true, afrag[mt], true, bfrag[nt], acc[mt][nt], false, false);
        __syncthreads();
    }

    float xs = __uint_as_float(*amax_bits) * (1.0f / 128.0f);
    float sf = xs * y_scale[0];
    #pragma unroll
    for (int mt = 0; mt < 2; ++mt)
        #pragma unroll
        for (int nt = 0; nt < 2; ++nt) {
            int n = block_n + wn * 32 + nt * 16 + l15;
            float bv = bias[n];
            #pragma unroll
            for (int r = 0; r < 8; ++r) {
                int m = block_m + wm * 32 + mt * 16 + half * 8 + r;
                Y[(size_t)m * N_DIM + n] = (float)acc[mt][nt][r] * sf + bv;
            }
        }
}

extern "C" void kernel_launch(void* const* d_in, const int* in_sizes, int n_in,
                              void* d_out, int out_size, void* d_ws, size_t ws_size,
                              hipStream_t stream) {
    const float*       x       = (const float*)d_in[0];
    const signed char* w_q     = (const signed char*)d_in[1];
    const float*       bias    = (const float*)d_in[2];
    const float*       y_scale = (const float*)d_in[3];
    float*             y       = (float*)d_out;

    unsigned int* amax = (unsigned int*)d_ws;
    signed char*  xq   = (signed char*)d_ws + 256;
    signed char*  wt   = xq + (size_t)M_DIM * K_DIM;

    const int n = M_DIM * K_DIM;
    const size_t need = 256 + (size_t)M_DIM * K_DIM + (size_t)N_DIM * K_DIM;

    hipMemsetAsync(d_ws, 0, 4, stream);
    absmax_kernel<<<2048, 256, 0, stream>>>(x, amax, n);
    quant_kernel <<<2048, 256, 0, stream>>>(x, (int*)xq, amax, n);

    if (ws_size >= need) {
        dim3 tgrid(N_DIM / 128, K_DIM / 64);
        wtrans_kernel<<<tgrid, 256, 0, stream>>>(w_q, wt);
        dim3 grid(N_DIM / BN, M_DIM / BM);
        gemm_i8_t_kernel<<<grid, 256, 0, stream>>>(xq, wt, bias, y_scale, amax, y);
    } else {
        dim3 grid(N_DIM / FBN, M_DIM / FBM);
        gemm_i8_kernel<<<grid, 256, 0, stream>>>(xq, w_q, bias, y_scale, amax, y);
    }
}
